// FaceAttnProcessor_57939108823386
// MI455X (gfx1250) — matrix operand order repaired and checked by reading, not compiled, not run on hardware
//
#include <hip/hip_runtime.h>

// ---------------------------------------------------------------------------
// Types for CDNA5 WMMA
// ---------------------------------------------------------------------------
typedef __attribute__((ext_vector_type(16))) __bf16          v16bf;
typedef __attribute__((ext_vector_type(16))) unsigned short  v16us;
typedef __attribute__((ext_vector_type(8)))  float           v8f;

struct U4 { unsigned int x, y, z, w; };   // POD 16B chunk (8 bf16)

#define WMMA_BF16(a, b, c) \
  __builtin_amdgcn_wmma_f32_16x16x32_bf16(false, (a), false, (b), (short)0, (c), false, false)

__device__ __forceinline__ unsigned short f2bf(float f) {
  unsigned int u = __builtin_bit_cast(unsigned int, f);
  unsigned int r = u + 0x7FFFu + ((u >> 16) & 1u);   // round-to-nearest-even
  return (unsigned short)(r >> 16);
}

__device__ __forceinline__ unsigned long long pack4bf(float4 f) {
  return (unsigned long long)f2bf(f.x)
       | ((unsigned long long)f2bf(f.y) << 16)
       | ((unsigned long long)f2bf(f.z) << 32)
       | ((unsigned long long)f2bf(f.w) << 48);
}

__device__ __forceinline__ v16bf us_to_bf(v16us t) {
  union { v16us us; v16bf bf; } cv; cv.us = t; return cv.bf;
}

// A-fragment (16x32 bf16): lane m = lane&15, half = lane>>4.
// VGPR v holds K pair: k = ((v&4)?16:0) + 2*(v&3) + 8*half
__device__ __forceinline__ v16bf ldfragA(const unsigned short* rowbase, int hgrp) {
  v16us t;
#pragma unroll
  for (int v = 0; v < 8; ++v) {
    int kk = ((v & 4) ? 16 : 0) + ((v & 3) << 1) + (hgrp << 3);
    unsigned int w = *(const unsigned int*)(rowbase + kk);
    t[2 * v]     = (unsigned short)(w & 0xffffu);
    t[2 * v + 1] = (unsigned short)(w >> 16);
  }
  return us_to_bf(t);
}

// B-fragment (32x16 bf16) from [n][k]-major LDS: lane n = lane&15.
// VGPR v holds K pair: k = ((v&4)?16:0) + 2*(v&3) + 8*half
__device__ __forceinline__ v16bf ldfragB(const unsigned short* rowbase, int hgrp) {
  v16us t;
#pragma unroll
  for (int v = 0; v < 8; ++v) {
    int kk = ((v & 4) ? 16 : 0) + ((v & 3) << 1) + (hgrp << 3);
    unsigned int w = *(const unsigned int*)(rowbase + kk);
    t[2 * v]     = (unsigned short)(w & 0xffffu);
    t[2 * v + 1] = (unsigned short)(w >> 16);
  }
  return us_to_bf(t);
}

// ---------------------------------------------------------------------------
// GEMM: C[M,N] = A[M,K](f32) x B[K,N](f32, row stride ldb), bf16 WMMA, f32 acc
// Epilogue: +bias[col], *mul[row,col], +res[row,col]; output f32 (Cc) or bf16 (Ch)
// Block: 256 thr = 8 waves (4x2), block tile 128x64, K-step 32.
// ---------------------------------------------------------------------------
__global__ __launch_bounds__(256) void gemm_bf16_wmma_kernel(
    const float* __restrict__ A, const float* __restrict__ Bw,
    const float* __restrict__ bias, const float* __restrict__ res,
    const float* __restrict__ mul, float* __restrict__ Cc,
    unsigned short* __restrict__ Ch,
    int M, int N, int K, int ldb)
{
  __shared__ unsigned short As[128][40];   // [m][k], +8 pad
  __shared__ unsigned short Bs[64][40];    // [n][k] (transposed at stage), +8 pad

  const int tid  = threadIdx.x;
  const int wv   = tid >> 5;
  const int lane = tid & 31;
  const int wm   = wv >> 1;         // 0..3 -> M
  const int wn   = wv & 1;          // 0..1 -> N
  const int hgrp = lane >> 4;
  const int l16  = lane & 15;
  const int m0   = blockIdx.y * 128;
  const int n0   = blockIdx.x * 64;

  v8f acc00 = {}, acc01 = {}, acc10 = {}, acc11 = {};

  for (int k0 = 0; k0 < K; k0 += 32) {
    // stage A tile 128x32 (1024 float4 / 256 thr = 4 each)
#pragma unroll
    for (int it = 0; it < 4; ++it) {
      int e = tid + (it << 8);
      int r = e >> 3;
      int c = (e & 7) << 2;
      float4 f = make_float4(0.f, 0.f, 0.f, 0.f);
      int grow = m0 + r;
      if (grow < M) f = *(const float4*)(A + (size_t)grow * K + k0 + c);
      *(unsigned long long*)&As[r][c] = pack4bf(f);
    }
    // stage B tile 32x64 transposed into [n][k]
#pragma unroll
    for (int it = 0; it < 2; ++it) {
      int e = tid + (it << 8);
      int r = e >> 4;
      int c = (e & 15) << 2;
      float4 f = *(const float4*)(Bw + (size_t)(k0 + r) * ldb + n0 + c);
      Bs[c + 0][r] = f2bf(f.x);
      Bs[c + 1][r] = f2bf(f.y);
      Bs[c + 2][r] = f2bf(f.z);
      Bs[c + 3][r] = f2bf(f.w);
    }
    // prefetch next weight tile (global_prefetch_b8 path)
    if (k0 + 32 < K)
      __builtin_prefetch(Bw + (size_t)(k0 + 32 + (tid >> 4)) * ldb + n0, 0, 0);
    __syncthreads();

    v16bf a0 = ldfragA(&As[wm * 32 + 0  + l16][0], hgrp);
    v16bf a1 = ldfragA(&As[wm * 32 + 16 + l16][0], hgrp);
    v16bf b0 = ldfragB(&Bs[wn * 32 + 0  + l16][0], hgrp);
    v16bf b1 = ldfragB(&Bs[wn * 32 + 16 + l16][0], hgrp);

    acc00 = WMMA_BF16(a0, b0, acc00);
    acc01 = WMMA_BF16(a0, b1, acc01);
    acc10 = WMMA_BF16(a1, b0, acc10);
    acc11 = WMMA_BF16(a1, b1, acc11);
    __syncthreads();
  }

  // epilogue: C layout (m = r + 8*half, n = lane&15)
#pragma unroll
  for (int tm = 0; tm < 2; ++tm) {
#pragma unroll
    for (int tn = 0; tn < 2; ++tn) {
      v8f acc = (tm == 0) ? (tn == 0 ? acc00 : acc01) : (tn == 0 ? acc10 : acc11);
#pragma unroll
      for (int r = 0; r < 8; ++r) {
        int row = m0 + wm * 32 + tm * 16 + r + (hgrp << 3);
        int col = n0 + wn * 32 + tn * 16 + l16;
        if (row < M && col < N) {
          float vv = acc[r];
          if (bias) vv += bias[col];
          if (mul)  vv *= mul[(size_t)row * N + col];
          if (res)  vv += res[(size_t)row * N + col];
          if (Ch) Ch[(size_t)row * N + col] = f2bf(vv);
          else    Cc[(size_t)row * N + col] = vv;
        }
      }
    }
  }
}

// ---------------------------------------------------------------------------
// Flash attention (head dim 64) over bf16 Q/K/V, bf16 WMMA, online softmax.
// Block = 128 thr = 4 waves; each wave owns 16 q rows (block: 64 q rows).
// Key tiles of 32. K staged via global_load_async_to_lds (ASYNCcnt path).
// grid = (Nq/64, H, B)
// ---------------------------------------------------------------------------
__global__ __launch_bounds__(128) void attn_wmma_kernel(
    const unsigned short* __restrict__ Q, const unsigned short* __restrict__ K,
    const unsigned short* __restrict__ V, float* __restrict__ O,
    int Nk, int qRPB, int kRPB, int oRPB, int C, float scale)
{
  __shared__ unsigned short Ks[32][72];      // [key][d] (row 144B, 16B aligned)
  __shared__ unsigned short Vs[64][40];      // [d][key] (transposed)
  __shared__ unsigned short Ps[4][16][40];   // per-wave P tile [m][key]

  const int b    = blockIdx.z;
  const int h    = blockIdx.y;
  const int tid  = threadIdx.x;
  const int wv   = tid >> 5;
  const int lane = tid & 31;
  const int hgrp = lane >> 4;
  const int l16  = lane & 15;
  const int qtile = blockIdx.x * 64 + wv * 16;

  const unsigned short* Qb = Q + ((size_t)b * qRPB) * C + (size_t)h * 64;
  const unsigned short* Kb = K + ((size_t)b * kRPB) * C + (size_t)h * 64;
  const unsigned short* Vb = V + ((size_t)b * kRPB) * C + (size_t)h * 64;
  float*                Ob = O + ((size_t)b * oRPB) * C + (size_t)h * 64;

  // Q fragments: bf16 A-layout is two contiguous 16B chunks per lane.
  v16bf qa[2];
#pragma unroll
  for (int ds = 0; ds < 2; ++ds) {
    const unsigned short* qrow = Qb + (size_t)(qtile + l16) * C + ds * 32;
    union { U4 c[2]; v16us u; } cv;
    cv.c[0] = *(const U4*)(qrow + (hgrp << 3));
    cv.c[1] = *(const U4*)(qrow + 16 + (hgrp << 3));
    qa[ds] = us_to_bf(cv.u);
  }

  v8f o0 = {}, o1 = {}, o2 = {}, o3 = {};
  float mrow[8], lrow[8];
#pragma unroll
  for (int r = 0; r < 8; ++r) { mrow[r] = -3.0e38f; lrow[r] = 0.f; }

  const int nkt = (Nk + 31) / 32;
  for (int kt = 0; kt < nkt; ++kt) {
    const int kb = kt * 32;
    // stage K tile 32x64 bf16 via async copy to LDS (16B per lane-transfer),
    // and V tile transposed into [d][key] via manual loads.
    // OOB key rows: clamp source address (finite data, masked in softmax).
#pragma unroll
    for (int it = 0; it < 2; ++it) {
      int ci  = tid + it * 128;          // 0..255 16B-chunks
      int row = ci >> 3;                 // key row 0..31
      int c8  = (ci & 7) << 3;           // d element offset 0..56
      int krow = kb + row; if (krow >= Nk) krow = Nk - 1;
      const unsigned short* gk = Kb + (size_t)krow * C + c8;
      unsigned loff = (unsigned)(size_t)(&Ks[row][c8]);
      asm volatile("global_load_async_to_lds_b128 %0, %1, off"
                   :: "v"(loff), "v"(gk) : "memory");
      U4 fv = *(const U4*)(Vb + (size_t)krow * C + c8);
      Vs[c8 + 0][row] = (unsigned short)(fv.x & 0xffffu);
      Vs[c8 + 1][row] = (unsigned short)(fv.x >> 16);
      Vs[c8 + 2][row] = (unsigned short)(fv.y & 0xffffu);
      Vs[c8 + 3][row] = (unsigned short)(fv.y >> 16);
      Vs[c8 + 4][row] = (unsigned short)(fv.z & 0xffffu);
      Vs[c8 + 5][row] = (unsigned short)(fv.z >> 16);
      Vs[c8 + 6][row] = (unsigned short)(fv.w & 0xffffu);
      Vs[c8 + 7][row] = (unsigned short)(fv.w >> 16);
    }
    asm volatile("s_wait_asynccnt 0x0" ::: "memory");
    __syncthreads();

    // S = Q K^T : two 16x16 n-subtiles, accumulate over d in two 32-chunks
    v8f s0 = {}, s1 = {};
#pragma unroll
    for (int ds = 0; ds < 2; ++ds) {
      v16bf bk0 = ldfragB(&Ks[0  + l16][ds * 32], hgrp);
      v16bf bk1 = ldfragB(&Ks[16 + l16][ds * 32], hgrp);
      s0 = WMMA_BF16(qa[ds], bk0, s0);
      s1 = WMMA_BF16(qa[ds], bk1, s1);
    }

    // online softmax (rows spread across 16-lane groups in C layout)
    const bool ok0 = (kb + l16) < Nk;
    const bool ok1 = (kb + 16 + l16) < Nk;
#pragma unroll
    for (int r = 0; r < 8; ++r) {
      float e0 = ok0 ? s0[r] * scale : -3.0e38f;
      float e1 = ok1 ? s1[r] * scale : -3.0e38f;
      float mx = fmaxf(e0, e1);
#pragma unroll
      for (int m = 8; m >= 1; m >>= 1) mx = fmaxf(mx, __shfl_xor(mx, m, 32));
      float mn = fmaxf(mrow[r], mx);
      float al = __expf(mrow[r] - mn);
      float p0 = __expf(e0 - mn);
      float p1 = __expf(e1 - mn);
      float ss = p0 + p1;
#pragma unroll
      for (int m = 8; m >= 1; m >>= 1) ss += __shfl_xor(ss, m, 32);
      lrow[r] = lrow[r] * al + ss;
      mrow[r] = mn;
      o0[r] *= al; o1[r] *= al; o2[r] *= al; o3[r] *= al;
      int mi = r + (hgrp << 3);
      Ps[wv][mi][l16]      = f2bf(p0);
      Ps[wv][mi][16 + l16] = f2bf(p1);
    }
    // wave-local LDS RAW fence before re-reading P in A-layout
    __builtin_amdgcn_wave_barrier();
    asm volatile("s_wait_dscnt 0x0" ::: "memory");
    __builtin_amdgcn_wave_barrier();

    v16bf pa = ldfragA(&Ps[wv][l16][0], hgrp);
    o0 = WMMA_BF16(pa, ldfragB(&Vs[0  + l16][0], hgrp), o0);
    o1 = WMMA_BF16(pa, ldfragB(&Vs[16 + l16][0], hgrp), o1);
    o2 = WMMA_BF16(pa, ldfragB(&Vs[32 + l16][0], hgrp), o2);
    o3 = WMMA_BF16(pa, ldfragB(&Vs[48 + l16][0], hgrp), o3);
    __syncthreads();
  }

#pragma unroll
  for (int r = 0; r < 8; ++r) {
    float inv = 1.0f / lrow[r];
    float* orow = Ob + (size_t)(qtile + r + (hgrp << 3)) * C;
    orow[0  + l16] = o0[r] * inv;
    orow[16 + l16] = o1[r] * inv;
    orow[32 + l16] = o2[r] * inv;
    orow[48 + l16] = o3[r] * inv;
  }
}

// ---------------------------------------------------------------------------
// LayerNorm: one block (256 thr) per row; out row remapped per batch.
// ---------------------------------------------------------------------------
__global__ __launch_bounds__(256) void layernorm_kernel(
    const float* __restrict__ in, float* __restrict__ out,
    const float* __restrict__ g, const float* __restrict__ bt,
    int rpbIn, int rpbOut, int C)
{
  int r = blockIdx.x;
  const float* x = in + (size_t)r * C;
  int b = r / rpbIn, i = r - b * rpbIn;
  float* y = out + ((size_t)b * rpbOut + i) * C;

  float s = 0.f, s2 = 0.f;
  for (int c = threadIdx.x; c < C; c += 256) { float v = x[c]; s += v; s2 += v * v; }
#pragma unroll
  for (int m = 16; m >= 1; m >>= 1) { s += __shfl_xor(s, m, 32); s2 += __shfl_xor(s2, m, 32); }
  __shared__ float rs[8], rq[8];
  int wv = threadIdx.x >> 5, lane = threadIdx.x & 31;
  if (lane == 0) { rs[wv] = s; rq[wv] = s2; }
  __syncthreads();
  s = 0.f; s2 = 0.f;
#pragma unroll
  for (int w = 0; w < 8; ++w) { s += rs[w]; s2 += rq[w]; }
  float mean = s / C;
  float var  = s2 / C - mean * mean;
  float inv  = rsqrtf(var + 1e-5f);
  for (int c = threadIdx.x; c < C; c += 256)
    y[c] = (x[c] - mean) * inv * g[c] + bt[c];
}

// ---------------------------------------------------------------------------
// Elementwise helpers
// ---------------------------------------------------------------------------
__global__ void residual_gate_kernel(const float* __restrict__ base,
                                     const float* __restrict__ delta,
                                     const float* __restrict__ alpha,
                                     float* __restrict__ out, long long n)
{
  long long i = (long long)blockIdx.x * blockDim.x + threadIdx.x;
  if (i < n) out[i] = base[i] + tanhf(alpha[0]) * delta[i];
}

__global__ void gelu_inplace_kernel(float* __restrict__ g, long long n)
{
  long long i = (long long)blockIdx.x * blockDim.x + threadIdx.x;
  if (i < n) { float x = g[i]; g[i] = 0.5f * x * (1.f + erff(x * 0.70710678f)); }
}

__global__ void gather_rows_kernel(const float* __restrict__ in, float* __restrict__ out,
                                   int rpb, int inStride, int inOff, int C)
{
  int r = blockIdx.x;
  int b = r / rpb, i = r - b * rpb;
  const float* src = in + ((size_t)b * inStride + inOff + i) * C;
  float* dst = out + (size_t)r * C;
  for (int c = threadIdx.x; c < C; c += blockDim.x) dst[c] = src[c];
}

// ---------------------------------------------------------------------------
// Orchestration
// ---------------------------------------------------------------------------
extern "C" void kernel_launch(void* const* d_in, const int* in_sizes, int n_in,
                              void* d_out, int out_size, void* d_ws, size_t ws_size,
                              hipStream_t stream)
{
  (void)in_sizes; (void)n_in; (void)out_size; (void)ws_size;

  const float* hidden   = (const float*)d_in[0];
  const float* enc      = (const float*)d_in[1];
  const float* ln1_g    = (const float*)d_in[2];
  const float* ln1_b    = (const float*)d_in[3];
  const float* ln2_g    = (const float*)d_in[4];
  const float* ln2_b    = (const float*)d_in[5];
  const float* sa_wq    = (const float*)d_in[6];
  const float* sa_wk    = (const float*)d_in[7];
  const float* sa_wv    = (const float*)d_in[8];
  const float* sa_wo    = (const float*)d_in[9];
  const float* sa_wo_b  = (const float*)d_in[10];
  const float* ff_ln_g  = (const float*)d_in[11];
  const float* ff_ln_b  = (const float*)d_in[12];
  const float* ff_w1    = (const float*)d_in[13];
  const float* ff_w2    = (const float*)d_in[14];
  const float* a_attn   = (const float*)d_in[15];
  const float* a_dense  = (const float*)d_in[16];
  const float* ca_wq    = (const float*)d_in[17];
  const float* ca_wk    = (const float*)d_in[18];
  const float* ca_wv    = (const float*)d_in[19];
  const float* ca_wo    = (const float*)d_in[20];
  const float* ca_wo_b  = (const float*)d_in[21];
  float* outp = (float*)d_out;

  const int B = 2, N = 1024, C = 768, Lenc = 93, T = 77, Fc = 16, NC = 1040;
  const int H = 12, INNER = 3072;
  const float scale = 0.125f; // d^-0.5, d=64

  float* ws = (float*)d_ws;
  size_t o = 0;
  float* xbuf = ws + o; o += (size_t)B * N * C;
  float* face = ws + o; o += (size_t)B * Fc * C;
  float* comb = ws + o; o += (size_t)B * NC * C;
  unsigned short* qb = (unsigned short*)(ws + o); o += (size_t)B * NC * C / 2;
  unsigned short* kb = (unsigned short*)(ws + o); o += (size_t)B * NC * C / 2;
  unsigned short* vb = (unsigned short*)(ws + o); o += (size_t)B * NC * C / 2;
  float* attO = ws + o; o += (size_t)B * N * C;
  float* proj = ws + o; o += (size_t)B * N * C;
  float* t1   = ws + o; o += (size_t)B * N * C;
  float* t2   = ws + o; o += (size_t)B * N * C;
  float* gate = ws + o; o += (size_t)B * N * INNER;
  float* ffg  = ws + o; o += (size_t)B * N * INNER;
  float* ffo  = ws + o; o += (size_t)B * N * C;
  unsigned short* caq = (unsigned short*)(ws + o); o += (size_t)B * N * C / 2;
  float* txt  = ws + o; o += (size_t)B * T * C;
  unsigned short* cak = (unsigned short*)(ws + o); o += (size_t)B * T * C / 2;
  unsigned short* cav = (unsigned short*)(ws + o); o += (size_t)B * T * C / 2;
  float* cao  = ws + o; o += (size_t)B * N * C;

  const long long nBNC = (long long)B * N * C;

  // 1) slice face tokens, layernorm x & face into comb (concat layout)
  gather_rows_kernel<<<B * Fc, 256, 0, stream>>>(enc, face, Fc, Lenc, T, C);
  layernorm_kernel<<<B * N, 256, 0, stream>>>(hidden, comb, ln1_g, ln1_b, N, NC, C);
  layernorm_kernel<<<B * Fc, 256, 0, stream>>>(face, comb + (size_t)N * C, ln1_g, ln1_b, Fc, NC, C);

  // 2) self-attention QKV GEMMs (M = B*1040), bf16 outputs for attention
  dim3 gQKV(C / 64, (B * NC + 127) / 128);
  gemm_bf16_wmma_kernel<<<gQKV, 256, 0, stream>>>(comb, sa_wq, nullptr, nullptr, nullptr, nullptr, qb, B * NC, C, C, C);
  gemm_bf16_wmma_kernel<<<gQKV, 256, 0, stream>>>(comb, sa_wk, nullptr, nullptr, nullptr, nullptr, kb, B * NC, C, C, C);
  gemm_bf16_wmma_kernel<<<gQKV, 256, 0, stream>>>(comb, sa_wv, nullptr, nullptr, nullptr, nullptr, vb, B * NC, C, C, C);

  // 3) self-attention (only first N q-rows needed; keys = 1040)
  attn_wmma_kernel<<<dim3(N / 64, H, B), 128, 0, stream>>>(qb, kb, vb, attO, NC, NC, NC, N, C, scale);

  // 4) out projection + gated residual
  dim3 gN(C / 64, (B * N + 127) / 128);
  gemm_bf16_wmma_kernel<<<gN, 256, 0, stream>>>(attO, sa_wo, sa_wo_b, nullptr, nullptr, proj, nullptr, B * N, C, C, C);
  residual_gate_kernel<<<(unsigned)((nBNC + 255) / 256), 256, 0, stream>>>(hidden, proj, a_attn, xbuf, nBNC);

  // 5) double layernorm + GEGLU FFN
  layernorm_kernel<<<B * N, 256, 0, stream>>>(xbuf, t1, ln2_g, ln2_b, N, N, C);
  layernorm_kernel<<<B * N, 256, 0, stream>>>(t1, t2, ff_ln_g, ff_ln_b, N, N, C);
  dim3 gFF(INNER / 64, (B * N + 127) / 128);
  // gate half: t2 @ ff_w1[:, INNER:]
  gemm_bf16_wmma_kernel<<<gFF, 256, 0, stream>>>(t2, ff_w1 + INNER, nullptr, nullptr, nullptr, gate, nullptr, B * N, INNER, C, 2 * INNER);
  gelu_inplace_kernel<<<(unsigned)(((long long)B * N * INNER + 255) / 256), 256, 0, stream>>>(gate, (long long)B * N * INNER);
  // a half fused with gate multiply: (t2 @ ff_w1[:, :INNER]) * gelu(gate)
  gemm_bf16_wmma_kernel<<<gFF, 256, 0, stream>>>(t2, ff_w1, nullptr, nullptr, gate, ffg, nullptr, B * N, INNER, C, 2 * INNER);
  gemm_bf16_wmma_kernel<<<gN, 256, 0, stream>>>(ffg, ff_w2, nullptr, nullptr, nullptr, ffo, nullptr, B * N, C, INNER, C);
  residual_gate_kernel<<<(unsigned)((nBNC + 255) / 256), 256, 0, stream>>>(xbuf, ffo, a_dense, xbuf, nBNC);

  // 6) cross-attention (bf16 Q/K/V)
  gemm_bf16_wmma_kernel<<<gN, 256, 0, stream>>>(xbuf, ca_wq, nullptr, nullptr, nullptr, nullptr, caq, B * N, C, C, C);
  gather_rows_kernel<<<B * T, 256, 0, stream>>>(enc, txt, T, Lenc, 0, C);
  dim3 gT(C / 64, (B * T + 127) / 128);
  gemm_bf16_wmma_kernel<<<gT, 256, 0, stream>>>(txt, ca_wk, nullptr, nullptr, nullptr, nullptr, cak, B * T, C, C, C);
  gemm_bf16_wmma_kernel<<<gT, 256, 0, stream>>>(txt, ca_wv, nullptr, nullptr, nullptr, nullptr, cav, B * T, C, C, C);
  attn_wmma_kernel<<<dim3(N / 64, H, B), 128, 0, stream>>>(caq, cak, cav, cao, T, N, T, N, C, scale);

  // 7) CA out projection with fused bias + residual -> final output
  gemm_bf16_wmma_kernel<<<gN, 256, 0, stream>>>(cao, ca_wo, ca_wo_b, xbuf, nullptr, outp, nullptr, B * N, C, C, C);
}
